// TransformerBlock_8598524526941
// MI455X (gfx1250) — compile-verified, matrix operand order checked by reading, not run on hardware
//
#include <hip/hip_runtime.h>
#include <hip/hip_bf16.h>
#include <math.h>

// ---------------------------------------------------------------------------
// Types for CDNA5 WMMA (wave32): v_wmma_f32_16x16x32_bf16
// ---------------------------------------------------------------------------
typedef __bf16 bf16_t;
typedef __bf16 v16bf __attribute__((ext_vector_type(16)));
typedef __bf16 v8bf  __attribute__((ext_vector_type(8)));
typedef __bf16 v4bf  __attribute__((ext_vector_type(4)));
typedef float  v8f   __attribute__((ext_vector_type(8)));

constexpr int D_MODEL = 1024;
constexpr int D_FF    = 2816;
constexpr int NHEAD   = 16;
constexpr int DKH     = 64;      // head dim
constexpr int SEQ     = 2048;
constexpr int BATCH   = 2;
constexpr int MTOT    = BATCH * SEQ;   // 4096 rows

// ---------------------------------------------------------------------------
// CDNA5 async memory->LDS copy (ASYNCcnt path, ISA ch.10 / 15.18).
// VDST VGPR carries the LDS byte address; per the flat-aperture rules the low
// 32 bits of a generic LDS pointer ARE the LDS offset, so truncation is exact.
// ---------------------------------------------------------------------------
__device__ __forceinline__ uint32_t lds_u32(const void* p) {
    return (uint32_t)(uintptr_t)p;
}
__device__ __forceinline__ void async_copy_b128(uint32_t lds_addr, const void* gptr) {
    asm volatile("global_load_async_to_lds_b128 %0, %1, off"
                 :: "v"(lds_addr), "v"(gptr) : "memory");
}
__device__ __forceinline__ void wait_async0() {
    asm volatile("s_wait_asynccnt 0x0" ::: "memory");
}

// ---------------------------------------------------------------------------
// Fragment loader. LDS tile layout: row-major [rows][stride], K contiguous,
// stride a multiple of 8 bf16 so every lane's 8-element group is 16B aligned.
// Matches ISA 16-bit A(16x32)/B(32x16) layouts:
//   lane 0-15 : elements 0..7 = K0..7,  elements 8..15 = K16..23
//   lane 16-31: elements 0..7 = K8..15, elements 8..15 = K24..31
// ---------------------------------------------------------------------------
__device__ __forceinline__ v16bf load_frag(const bf16_t* base, int stride, int lane) {
    const int rn = lane & 15;
    const int kb = (lane >> 4) << 3;
    const bf16_t* p = base + rn * stride + kb;
    v8bf lo = *(const v8bf*)(p);
    v8bf hi = *(const v8bf*)(p + 16);
    v16bf r;
#pragma unroll
    for (int i = 0; i < 8; ++i) { r[i] = lo[i]; r[i + 8] = hi[i]; }
    return r;
}

__device__ __forceinline__ v8f wmma_bf16(v16bf a, v16bf b, v8f c) {
    return __builtin_amdgcn_wmma_f32_16x16x32_bf16(false, a, false, b, (short)0, c,
                                                   false, false);
}

__device__ __forceinline__ v8f zero_v8f() {
    v8f z;
#pragma unroll
    for (int i = 0; i < 8; ++i) z[i] = 0.0f;
    return z;
}

// ---------------------------------------------------------------------------
// fp32 -> bf16 bulk conversion (weights, one pass per launch).
// ---------------------------------------------------------------------------
__global__ __launch_bounds__(256) void cvt_bf16_kernel(const float* __restrict__ src,
                                                       bf16_t* __restrict__ dst) {
    const size_t i = ((size_t)blockIdx.x * 256 + threadIdx.x) * 4;
    const float4 f = *(const float4*)(src + i);
    v4bf t; t[0] = (bf16_t)f.x; t[1] = (bf16_t)f.y;
            t[2] = (bf16_t)f.z; t[3] = (bf16_t)f.w;
    *(v4bf*)(dst + i) = t;
}

// ---------------------------------------------------------------------------
// LayerNorm: one 256-thread block per row; fp32 in, bf16 out (GEMM A operand).
// ---------------------------------------------------------------------------
__global__ __launch_bounds__(256) void ln_kernel(const float* __restrict__ x,
                                                 const float* __restrict__ g,
                                                 const float* __restrict__ b,
                                                 bf16_t* __restrict__ y) {
    const int row = blockIdx.x;
    const float* xr = x + (size_t)row * D_MODEL;
    float s = 0.f, ss = 0.f;
#pragma unroll
    for (int i = 0; i < 4; ++i) {
        float v = xr[threadIdx.x + 256 * i];
        s += v; ss += v * v;
    }
#pragma unroll
    for (int off = 16; off; off >>= 1) {
        s  += __shfl_xor(s,  off, 32);
        ss += __shfl_xor(ss, off, 32);
    }
    __shared__ float rs[8], rss[8];
    const int wave = threadIdx.x >> 5, lane = threadIdx.x & 31;
    if (lane == 0) { rs[wave] = s; rss[wave] = ss; }
    __syncthreads();
    s = 0.f; ss = 0.f;
#pragma unroll
    for (int i = 0; i < 8; ++i) { s += rs[i]; ss += rss[i]; }
    const float mu  = s * (1.0f / D_MODEL);
    const float var = ss * (1.0f / D_MODEL) - mu * mu;
    const float inv = rsqrtf(var + 1e-5f);
    bf16_t* yr = y + (size_t)row * D_MODEL;
#pragma unroll
    for (int i = 0; i < 4; ++i) {
        int c = threadIdx.x + 256 * i;
        yr[c] = (bf16_t)((xr[c] - mu) * inv * g[c] + b[c]);
    }
}

// ---------------------------------------------------------------------------
// WMMA GEMM: C[M,N] = A[M,K](bf16) * W[N,K]^T(bf16) (+ epilogue, fp32 out)
//   EPI 0: none   1: +bias   2: silu(+bias)   3: +resid   4: +bias +resid
// Tile 128x128 per 256-thread block. K staged 32/iter via DOUBLE-BUFFERED
// async DMA: tile k+1 is in flight on ASYNCcnt while tile k runs on the XDL
// pipe; one barrier per K-step.
// ---------------------------------------------------------------------------
template <int EPI>
__global__ __launch_bounds__(256) void gemm_wmma_kernel(
    const bf16_t* __restrict__ A, const bf16_t* __restrict__ W,
    const float* __restrict__ bias, const float* __restrict__ resid,
    float* __restrict__ C, int M, int N, int K)
{
    __shared__ __align__(16) bf16_t lA[2][128 * 40];
    __shared__ __align__(16) bf16_t lB[2][128 * 40];
    const int tid  = threadIdx.x;
    const int lane = tid & 31;
    const int wave = tid >> 5;
    const int rowBlk = blockIdx.y * 128;
    const int colBlk = blockIdx.x * 128;
    const int wm = (wave >> 2) * 64;   // 0 / 64
    const int wn = (wave & 3) * 32;    // 0 / 32 / 64 / 96
    const uint32_t lAb[2] = { lds_u32(lA[0]), lds_u32(lA[1]) };
    const uint32_t lBb[2] = { lds_u32(lB[0]), lds_u32(lB[1]) };

    // Per-thread staging coordinates: 512 b128 chunks cover a 128x32 tile.
    const int sr0 = (tid      ) >> 2, sc0 = ((tid      ) & 3) << 3;
    const int sr1 = (tid + 256) >> 2, sc1 = ((tid + 256) & 3) << 3;

    auto stage = [&](int buf, int k0) {
        async_copy_b128(lAb[buf] + (uint32_t)(sr0 * 40 + sc0) * 2u,
                        A + (size_t)(rowBlk + sr0) * K + k0 + sc0);
        async_copy_b128(lBb[buf] + (uint32_t)(sr0 * 40 + sc0) * 2u,
                        W + (size_t)(colBlk + sr0) * K + k0 + sc0);
        async_copy_b128(lAb[buf] + (uint32_t)(sr1 * 40 + sc1) * 2u,
                        A + (size_t)(rowBlk + sr1) * K + k0 + sc1);
        async_copy_b128(lBb[buf] + (uint32_t)(sr1 * 40 + sc1) * 2u,
                        W + (size_t)(colBlk + sr1) * K + k0 + sc1);
    };

    v8f acc[4][2];
#pragma unroll
    for (int i = 0; i < 4; ++i)
#pragma unroll
        for (int j = 0; j < 2; ++j) acc[i][j] = zero_v8f();

    stage(0, 0);
    int cur = 0;
    for (int k0 = 0; k0 < K; k0 += 32) {
        wait_async0();       // tile for this step has landed in LDS
        __syncthreads();     // everyone done reading the buffer we overwrite next
        if (k0 + 32 < K) stage(cur ^ 1, k0 + 32);  // DMA overlaps the WMMAs below

        v16bf bfr0 = load_frag(&lB[cur][(wn +  0) * 40], 40, lane);
        v16bf bfr1 = load_frag(&lB[cur][(wn + 16) * 40], 40, lane);
#pragma unroll
        for (int mi = 0; mi < 4; ++mi) {
            v16bf af = load_frag(&lA[cur][(wm + mi * 16) * 40], 40, lane);
            acc[mi][0] = wmma_bf16(af, bfr0, acc[mi][0]);
            acc[mi][1] = wmma_bf16(af, bfr1, acc[mi][1]);
        }
        cur ^= 1;
    }

    // Epilogue + store. C layout: row = base + r + 8*(lane/16), col = base + lane%16.
    const int ln16 = lane & 15;
    const int mo   = (lane >> 4) << 3;
#pragma unroll
    for (int ni = 0; ni < 2; ++ni) {
        const int col = colBlk + wn + ni * 16 + ln16;
        float bv = 0.0f;
        if constexpr (EPI == 1 || EPI == 2 || EPI == 4) bv = bias[col];
#pragma unroll
        for (int mi = 0; mi < 4; ++mi) {
#pragma unroll
            for (int r = 0; r < 8; ++r) {
                const int row = rowBlk + wm + mi * 16 + mo + r;
                float v = acc[mi][ni][r] + bv;
                if constexpr (EPI == 2) v = v * (1.0f / (1.0f + __expf(-v)));  // SiLU
                if constexpr (EPI == 3 || EPI == 4) v += resid[(size_t)row * N + col];
                C[(size_t)row * N + col] = v;
            }
        }
    }
}

// ---------------------------------------------------------------------------
// RoPE + reshape to heads: src fp32 [M, D_MODEL] -> dst bf16 [B*H, S, 64].
// scale folds the 1/sqrt(d_k) softmax scaling into Q.
// ---------------------------------------------------------------------------
template <bool ROPE>
__global__ __launch_bounds__(256) void rope_kernel(const float* __restrict__ src,
                                                   bf16_t* __restrict__ dst,
                                                   float scale) {
    const int idx = blockIdx.x * 256 + threadIdx.x;  // M * H * 32
    const int i   = idx & 31;
    const int h   = (idx >> 5) & (NHEAD - 1);
    const int row = idx >> 9;
    const int s   = row & (SEQ - 1);
    const int b   = row >> 11;
    const float* p = src + (size_t)row * D_MODEL + h * DKH;
    float x0 = p[i], x1 = p[i + 32];
    float o0 = x0, o1 = x1;
    if (ROPE) {
        const float theta = __powf(10000.0f, -(float)i * (1.0f / 32.0f));
        const float ang = (float)s * theta;
        const float c = __cosf(ang), sn = __sinf(ang);
        o0 = x0 * c - x1 * sn;
        o1 = x1 * c + x0 * sn;
    }
    bf16_t* q = dst + (((size_t)(b * NHEAD + h)) * SEQ + s) * DKH;
    q[i]      = (bf16_t)(o0 * scale);
    q[i + 32] = (bf16_t)(o1 * scale);
}

// ---------------------------------------------------------------------------
// Fused attention (flash style). grid = (S/128, B*H). Each wave owns 16 query
// rows. 32-key tiles are DOUBLE-BUFFERED: async-DMA K tile + manual V
// transpose for step j+1 run while step j's QK^T / softmax / P*V execute.
// One barrier per key step. Output written as bf16.
// ---------------------------------------------------------------------------
__global__ __launch_bounds__(256) void attn_kernel(
    const bf16_t* __restrict__ qb, const bf16_t* __restrict__ kb,
    const bf16_t* __restrict__ vb, const int* __restrict__ mask,
    bf16_t* __restrict__ mh)
{
    __shared__ __align__(16) bf16_t lK [2][32 * 72];   // K tile, row-major (keys x 64)
    __shared__ __align__(16) bf16_t lVt[2][64 * 40];   // V tile, transposed (d x 32)
    __shared__ __align__(16) bf16_t lP [8][16 * 40];   // per-wave P (C->A transpose)

    const int tid  = threadIdx.x;
    const int lane = tid & 31;
    const int wave = tid >> 5;
    const int bh = blockIdx.y, b = bh >> 4, h = bh & 15;
    const int qRow = blockIdx.x * 128 + wave * 16;

    const int ln16 = lane & 15;
    const int kb8  = (lane >> 4) << 3;
    const int mo   = (lane >> 4) << 3;
    const uint32_t lKb[2] = { lds_u32(lK[0]), lds_u32(lK[1]) };

    // Staging coordinates: 256 b128 chunks cover a 32x64 tile.
    const int sr = tid >> 3;            // key row 0..31
    const int sc = (tid & 7) << 3;      // dim 0,8,..,56

    auto stage = [&](int buf, int j) {
        async_copy_b128(lKb[buf] + (uint32_t)(sr * 72 + sc) * 2u,
                        kb + ((size_t)bh * SEQ + j + sr) * DKH + sc);
        v8bf vv = *(const v8bf*)(vb + ((size_t)bh * SEQ + j + sr) * DKH + sc);
#pragma unroll
        for (int e = 0; e < 8; ++e) lVt[buf][(sc + e) * 40 + sr] = vv[e];
    };

    // Q fragments (A layout) straight from global: row = qRow + lane%16.
    const bf16_t* qrow = qb + ((size_t)bh * SEQ + qRow + ln16) * DKH;
    v16bf qf[2];
#pragma unroll
    for (int ks = 0; ks < 2; ++ks) {
        v8bf lo = *(const v8bf*)(qrow + ks * 32 + kb8);
        v8bf hi = *(const v8bf*)(qrow + ks * 32 + kb8 + 16);
#pragma unroll
        for (int i = 0; i < 8; ++i) { qf[ks][i] = lo[i]; qf[ks][i + 8] = hi[i]; }
    }

    v8f o[4];
#pragma unroll
    for (int i = 0; i < 4; ++i) o[i] = zero_v8f();
    float mrun[8], lrun[8];
#pragma unroll
    for (int r = 0; r < 8; ++r) { mrun[r] = -1e30f; lrun[r] = 0.0f; }

    stage(0, 0);
    int cur = 0;
    for (int j = 0; j < SEQ; j += 32) {
        wait_async0();
        __syncthreads();
        if (j + 32 < SEQ) stage(cur ^ 1, j + 32);   // overlap DMA with compute

        // S = Q(16x64) * K^T(64x32): two 16-col subtiles, two K steps.
        v8f s0 = zero_v8f(), s1 = zero_v8f();
#pragma unroll
        for (int ks = 0; ks < 2; ++ks) {
            s0 = wmma_bf16(qf[ks], load_frag(&lK[cur][ks * 32],           72, lane), s0);
            s1 = wmma_bf16(qf[ks], load_frag(&lK[cur][16 * 72 + ks * 32], 72, lane), s1);
        }

        const int mk0 = mask[b * SEQ + j + ln16];
        const int mk1 = mask[b * SEQ + j + 16 + ln16];

        // Online softmax. Row m = mo + r lives on one 16-lane half.
#pragma unroll
        for (int r = 0; r < 8; ++r) {
            float v0 = mk0 ? s0[r] : -1e9f;
            float v1 = mk1 ? s1[r] : -1e9f;
            float rm = fmaxf(v0, v1);
#pragma unroll
            for (int off = 8; off; off >>= 1) rm = fmaxf(rm, __shfl_xor(rm, off, 16));
            const float mnew  = fmaxf(mrun[r], rm);
            const float alpha = __expf(mrun[r] - mnew);
            const float p0 = __expf(v0 - mnew);
            const float p1 = __expf(v1 - mnew);
            float ps = p0 + p1;
#pragma unroll
            for (int off = 8; off; off >>= 1) ps += __shfl_xor(ps, off, 16);
            lrun[r] = lrun[r] * alpha + ps;
            mrun[r] = mnew;
#pragma unroll
            for (int nn = 0; nn < 4; ++nn) o[nn][r] *= alpha;
            lP[wave][(mo + r) * 40 + ln16]      = (bf16_t)p0;
            lP[wave][(mo + r) * 40 + 16 + ln16] = (bf16_t)p1;
        }

        // O += P(16x32) * V(32x64). LDS ops within a wave are in-order.
        v16bf pf = load_frag(&lP[wave][0], 40, lane);
#pragma unroll
        for (int nn = 0; nn < 4; ++nn) {
            v16bf vf = load_frag(&lVt[cur][(nn * 16) * 40], 40, lane);
            o[nn] = wmma_bf16(pf, vf, o[nn]);
        }
        cur ^= 1;
    }

    // Normalize and scatter back to [B, S, D_MODEL] (bf16) with head interleave.
#pragma unroll
    for (int r = 0; r < 8; ++r) {
        const float inv = 1.0f / lrun[r];
        const int qi = qRow + mo + r;
        const size_t base = ((size_t)b * SEQ + qi) * D_MODEL + h * DKH;
#pragma unroll
        for (int nn = 0; nn < 4; ++nn)
            mh[base + nn * 16 + ln16] = (bf16_t)(o[nn][r] * inv);
    }
}

// ---------------------------------------------------------------------------
// Elementwise h = gate * up, fp32 in -> bf16 out (A operand of down-proj).
// ---------------------------------------------------------------------------
__global__ __launch_bounds__(256) void mul_kernel(const float* __restrict__ a,
                                                  const float* __restrict__ b,
                                                  bf16_t* __restrict__ c) {
    const size_t i = ((size_t)blockIdx.x * 256 + threadIdx.x) * 4;
    const float4 x = *(const float4*)(a + i);
    const float4 y = *(const float4*)(b + i);
    v4bf z; z[0] = (bf16_t)(x.x * y.x); z[1] = (bf16_t)(x.y * y.y);
            z[2] = (bf16_t)(x.z * y.z); z[3] = (bf16_t)(x.w * y.w);
    *(v4bf*)(c + i) = z;
}

// ---------------------------------------------------------------------------
// Host-side orchestration.
// ---------------------------------------------------------------------------
extern "C" void kernel_launch(void* const* d_in, const int* in_sizes, int n_in,
                              void* d_out, int out_size, void* d_ws, size_t ws_size,
                              hipStream_t stream) {
    (void)in_sizes; (void)n_in; (void)out_size; (void)ws_size;

    const float* x      = (const float*)d_in[0];
    const int*   mask   = (const int*)  d_in[1];
    const float* w_q    = (const float*)d_in[2];
    const float* w_k    = (const float*)d_in[3];
    const float* w_v    = (const float*)d_in[4];
    const float* w_o    = (const float*)d_in[5];
    const float* w_gate = (const float*)d_in[6];
    const float* b_gate = (const float*)d_in[7];
    const float* w_up   = (const float*)d_in[8];
    const float* b_up   = (const float*)d_in[9];
    const float* w_down = (const float*)d_in[10];
    const float* b_down = (const float*)d_in[11];
    const float* ln1_g  = (const float*)d_in[12];
    const float* ln1_b  = (const float*)d_in[13];
    const float* ln2_g  = (const float*)d_in[14];
    const float* ln2_b  = (const float*)d_in[15];

    char* ws = (char*)d_ws;
    size_t off = 0;
    auto alloc = [&](size_t bytes) -> void* {
        void* p = ws + off;
        off += (bytes + 255) & ~(size_t)255;
        return p;
    };
    const size_t MD  = (size_t)MTOT * D_MODEL;
    const size_t MF  = (size_t)MTOT * D_FF;
    const size_t DD  = (size_t)D_MODEL * D_MODEL;
    const size_t DF  = (size_t)D_FF * D_MODEL;

    bf16_t* xnb  = (bf16_t*)alloc(MD * 2);     // LN output (A operand), reused for LN2
    float*  q32  = (float*) alloc(MD * 4);
    float*  k32  = (float*) alloc(MD * 4);     // reused as x2
    float*  v32  = (float*) alloc(MD * 4);
    bf16_t* qb   = (bf16_t*)alloc(MD * 2);
    bf16_t* kbuf = (bf16_t*)alloc(MD * 2);
    bf16_t* vbuf = (bf16_t*)alloc(MD * 2);
    bf16_t* mhb  = (bf16_t*)alloc(MD * 2);
    float*  gate = (float*) alloc(MF * 4);
    float*  up   = (float*) alloc(MF * 4);
    bf16_t* hb   = (bf16_t*)alloc(MF * 2);
    bf16_t* wqb  = (bf16_t*)alloc(DD * 2);
    bf16_t* wkb  = (bf16_t*)alloc(DD * 2);
    bf16_t* wvb  = (bf16_t*)alloc(DD * 2);
    bf16_t* wob  = (bf16_t*)alloc(DD * 2);
    bf16_t* wgb  = (bf16_t*)alloc(DF * 2);
    bf16_t* wub  = (bf16_t*)alloc(DF * 2);
    bf16_t* wdb  = (bf16_t*)alloc(DF * 2);
    float*  x2   = k32;
    float*  out  = (float*)d_out;

    const dim3 blk(256);
    const dim3 gD(D_MODEL / 128, MTOT / 128);  // (8, 32)
    const dim3 gF(D_FF / 128,    MTOT / 128);  // (22, 32)
    const int ropeBlocks = (MTOT * NHEAD * 32) / 256;

    // One-shot weight conversion to bf16 (WMMA operand precision).
    cvt_bf16_kernel<<<DD / 1024, blk, 0, stream>>>(w_q,    wqb);
    cvt_bf16_kernel<<<DD / 1024, blk, 0, stream>>>(w_k,    wkb);
    cvt_bf16_kernel<<<DD / 1024, blk, 0, stream>>>(w_v,    wvb);
    cvt_bf16_kernel<<<DD / 1024, blk, 0, stream>>>(w_o,    wob);
    cvt_bf16_kernel<<<DF / 1024, blk, 0, stream>>>(w_gate, wgb);
    cvt_bf16_kernel<<<DF / 1024, blk, 0, stream>>>(w_up,   wub);
    cvt_bf16_kernel<<<DF / 1024, blk, 0, stream>>>(w_down, wdb);

    // LN1 -> Q/K/V projections
    ln_kernel<<<MTOT, blk, 0, stream>>>(x, ln1_g, ln1_b, xnb);
    gemm_wmma_kernel<0><<<gD, blk, 0, stream>>>(xnb, wqb, nullptr, nullptr, q32,
                                                MTOT, D_MODEL, D_MODEL);
    gemm_wmma_kernel<0><<<gD, blk, 0, stream>>>(xnb, wkb, nullptr, nullptr, k32,
                                                MTOT, D_MODEL, D_MODEL);
    gemm_wmma_kernel<0><<<gD, blk, 0, stream>>>(xnb, wvb, nullptr, nullptr, v32,
                                                MTOT, D_MODEL, D_MODEL);

    // RoPE + head reshape to bf16 (1/sqrt(d_k) folded into Q)
    rope_kernel<true ><<<ropeBlocks, blk, 0, stream>>>(q32, qb,   0.125f);
    rope_kernel<true ><<<ropeBlocks, blk, 0, stream>>>(k32, kbuf, 1.0f);
    rope_kernel<false><<<ropeBlocks, blk, 0, stream>>>(v32, vbuf, 1.0f);

    // Fused flash attention (bf16 output)
    attn_kernel<<<dim3(SEQ / 128, BATCH * NHEAD), blk, 0, stream>>>(
        qb, kbuf, vbuf, mask, mhb);

    // O projection + residual (x2 = x + mh @ w_o^T)
    gemm_wmma_kernel<3><<<gD, blk, 0, stream>>>(mhb, wob, nullptr, x, x2,
                                                MTOT, D_MODEL, D_MODEL);

    // LN2 -> SwiGLU FFN
    ln_kernel<<<MTOT, blk, 0, stream>>>(x2, ln2_g, ln2_b, xnb);
    gemm_wmma_kernel<2><<<gF, blk, 0, stream>>>(xnb, wgb, b_gate, nullptr, gate,
                                                MTOT, D_FF, D_MODEL);
    gemm_wmma_kernel<1><<<gF, blk, 0, stream>>>(xnb, wub, b_up, nullptr, up,
                                                MTOT, D_FF, D_MODEL);
    mul_kernel<<<MF / 1024, blk, 0, stream>>>(gate, up, hb);
    gemm_wmma_kernel<4><<<gD, blk, 0, stream>>>(hb, wdb, b_down, x2, out,
                                                MTOT, D_MODEL, D_FF);
}
